// WESN_1331439861781
// MI455X (gfx1250) — matrix operand order repaired and checked by reading, not compiled, not run on hardware
//
#include <hip/hip_runtime.h>

// ---------------- problem constants ----------------
#define BB   32          // batch
#define TT   512         // timesteps
#define DD   64          // input dim
#define UU   512         // reservoir units
#define WINW 4
#define RR   32          // readout
#define K1   (UU + DD)   // scan GEMM K = 576
#define KT1  (K1 / 32)   // 18 K-tiles for scan
#define FF   (UU + DD * WINW)   // 768
#define KTRO (FF / 32)   // 24 K-tiles for readout
#define NWG  8           // persistent workgroups in scan

// swizzled B-panel sizes (dwords): [mat][kt][ntile][lane][8]
#define NB_SCAN_DW (3 * KT1 * 32 * 32 * 8)        // 442368
#define NB_RO_DW   (3 * KTRO * 2 * 32 * 8)        // 36864
#define NX_ELEMS   ((TT + 3) * BB * DD)           // 1054720 bf16 (3 zero lag slots)
#define RES_ELEMS  ((TT + 1) * BB * UU)           // 8404992 bf16 per component

// dynamic LDS for the scan: hR (32KB) + hI (32KB) + x_t (4KB)
#define SMEM_BYTES ((2 * BB * UU + BB * DD) * 2)  // 69632

typedef __attribute__((ext_vector_type(16))) __bf16 v16bf;
typedef __attribute__((ext_vector_type(8)))  float  v8f;
typedef int v4i __attribute__((vector_size(16)));   // builtin's pointee type

struct U4 { unsigned x, y, z, w; };
union Tile { v16bf v; U4 q[2]; };

#if defined(__has_builtin)
#if __has_builtin(__builtin_amdgcn_global_load_async_to_lds_b128) && \
    __has_builtin(__builtin_amdgcn_s_wait_asynccnt)
#define WESN_ASYNC_LDS 1
#endif
#endif

static __device__ __forceinline__ unsigned short f2bf(float f) {
    union { float f; unsigned u; } v; v.f = f;
    unsigned r = v.u + 0x7FFFu + ((v.u >> 16) & 1u);   // round-to-nearest-even
    return (unsigned short)(r >> 16);
}
static __device__ __forceinline__ float bf2f(unsigned short h) {
    union { unsigned u; float f; } v; v.u = ((unsigned)h) << 16;
    return v.f;
}

// copy 16B global -> LDS; async (ASYNCcnt) when the builtin exists
static __device__ __forceinline__ void cp16_g2l(const unsigned short* g, unsigned short* l) {
#ifdef WESN_ASYNC_LDS
    __builtin_amdgcn_global_load_async_to_lds_b128((v4i*)g, (v4i*)l, 0, 0);
#else
    *(U4*)l = *(const U4*)g;
#endif
}
static __device__ __forceinline__ void cp_wait() {
#ifdef WESN_ASYNC_LDS
    __builtin_amdgcn_s_wait_asynccnt(0);
#endif
}

#define WMMA_BF16(A, B, C) \
    __builtin_amdgcn_wmma_f32_16x16x32_bf16(false, (A), false, (B), (short)0, (C), false, false)

// ---------------------------------------------------------------------------
// Prep 1: swizzle the scan B-panels into wave32 WMMA register order.
// B element (k, n):  k<512 -> W_rec^T[k][n] = rec[n][k] ;  k>=512 -> W_in[k-512][n]
// mats: 0 = real, 1 = imag, 2 = -imag (for accR's  Ai * (-Bi) term)
// B layout (16-bit, 32x16): lanes 0-15 hold K=0..7,16..23 (vj = K{2j,2j+1}, +8 for j>=4), lanes 16-31 hold K=8..15,24..31.
// ---------------------------------------------------------------------------
__global__ void wesn_prep_scanB(const float* __restrict__ rec_r, const float* __restrict__ rec_i,
                                const float* __restrict__ ker_r, const float* __restrict__ ker_i,
                                unsigned* __restrict__ dst) {
    int idx = blockIdx.x * blockDim.x + threadIdx.x;
    if (idx >= NB_SCAN_DW) return;
    int j    = idx & 7;
    int lane = (idx >> 3) & 31;
    int nt   = (idx >> 8) & 31;
    int q    = idx >> 13;
    int kt   = q % KT1;
    int mat  = q / KT1;
    int n    = nt * 16 + (lane & 15);
    int k0   = kt * 32 + ((lane & 16) >> 1) + 2 * j + ((j & 4) ? 8 : 0);
    float e[2];
#pragma unroll
    for (int s = 0; s < 2; ++s) {
        int k = k0 + s;
        float v;
        if (k < UU) v = (mat == 0 ? rec_r : rec_i)[n * UU + k];
        else        v = (mat == 0 ? ker_r : ker_i)[(k - UU) * UU + n];
        if (mat == 2) v = -v;
        e[s] = v;
    }
    dst[idx] = (unsigned)f2bf(e[0]) | ((unsigned)f2bf(e[1]) << 16);
}

// ---------------------------------------------------------------------------
// Prep 2: swizzle readout B-panels.  ro element (f, r) = readout[f*R + r].
// mats: 0 = readout_r, 1 = readout_i, 2 = -readout_i
// ---------------------------------------------------------------------------
__global__ void wesn_prep_roB(const float* __restrict__ ror, const float* __restrict__ roi,
                              unsigned* __restrict__ dst) {
    int idx = blockIdx.x * blockDim.x + threadIdx.x;
    if (idx >= NB_RO_DW) return;
    int j    = idx & 7;
    int lane = (idx >> 3) & 31;
    int nt   = (idx >> 8) & 1;
    int q    = idx >> 9;
    int kt   = q % KTRO;
    int mat  = q / KTRO;
    int r    = nt * 16 + (lane & 15);
    int k0   = kt * 32 + ((lane & 16) >> 1) + 2 * j + ((j & 4) ? 8 : 0);
    float e[2];
#pragma unroll
    for (int s = 0; s < 2; ++s) {
        int k = k0 + s;
        float v = (mat == 0 ? ror[k * RR + r] : roi[k * RR + r]);
        if (mat == 2) v = -v;
        e[s] = v;
    }
    dst[idx] = (unsigned)f2bf(e[0]) | ((unsigned)f2bf(e[1]) << 16);
}

// ---------------------------------------------------------------------------
// Prep 3: x -> bf16 [T+3][B][D] with 3 leading zero slots (lags), and zero h0.
// ---------------------------------------------------------------------------
__global__ void wesn_prep_x(const float* __restrict__ x, unsigned short* __restrict__ xbf,
                            unsigned short* __restrict__ resR, unsigned short* __restrict__ resI) {
    int idx = blockIdx.x * blockDim.x + threadIdx.x;
    if (idx < NX_ELEMS) {
        int d = idx & 63, b = (idx >> 6) & 31, slot = idx >> 11;
        float v = (slot < 3) ? 0.f : x[(b * TT + (slot - 3)) * DD + d];
        xbf[idx] = f2bf(v);
    } else {
        int r = idx - NX_ELEMS;
        if (r < BB * UU)          resR[r] = 0;
        else if (r < 2 * BB * UU) resI[r - BB * UU] = 0;
    }
}

// ---------------------------------------------------------------------------
// Scan: persistent, grid-synced over T steps.
// 8 WGs x 256 threads; wave w owns one (16x16) complex output tile.
// Per step: the full A panel (hR, hI, x_t = 68KB) is staged global->LDS with
// GLOBAL_LOAD_ASYNC_TO_LDS_B128 (shared by all 8 waves), then the K loop feeds
// WMMA from ds_load_b128 while the swizzled B panels are re-read from global memory.
// ---------------------------------------------------------------------------
__global__ void __launch_bounds__(256) wesn_scan(
    const unsigned* __restrict__ Bs, const unsigned short* __restrict__ xbf,
    unsigned short* __restrict__ resR, unsigned short* __restrict__ resI,
    const float* __restrict__ bias_r, const float* __restrict__ bias_i,
    unsigned* __restrict__ ctr) {
    extern __shared__ char smem[];
    unsigned short* shR = (unsigned short*)smem;         // [32][512]
    unsigned short* shI = shR + BB * UU;                 // [32][512]
    unsigned short* shX = shI + BB * UU;                 // [32][64]

    const int tid  = threadIdx.x;
    const int lane = tid & 31;
    const int wave = tid >> 5;
    const int mt   = wave & 1;
    const int ntg  = blockIdx.x * 4 + (wave >> 1);   // global n-tile 0..31
    const int col  = lane & 15;
    const int n    = ntg * 16 + col;
    const int hi8  = (lane & 16) >> 1;               // A-tile K offset for hi lanes
    const int mrow = mt * 16 + col;                  // A row for this lane
    const int mout = mt * 16 + ((lane & 16) ? 8 : 0);

    const float bvr = bias_r[n], bvi = bias_i[n];

    const U4* Br  = (const U4*)Bs;
    const U4* Bi  = Br + KT1 * 32 * 32 * 2;
    const U4* Bni = Bi + KT1 * 32 * 32 * 2;

    for (int t = 0; t < TT; ++t) {
        // ---- stage A panel (state + input) into LDS, async ----
        {
            const unsigned short* gR = resR + (size_t)t * (BB * UU);
            const unsigned short* gI = resI + (size_t)t * (BB * UU);
            const unsigned short* gX = xbf + (size_t)(t + 3) * (BB * DD);
#pragma unroll
            for (int i = 0; i < 8; ++i) {
                const int e = (i * 256 + tid) * 8;   // 16B chunk per thread
                cp16_g2l(gR + e, shR + e);
                cp16_g2l(gI + e, shI + e);
            }
            cp16_g2l(gX + tid * 8, shX + tid * 8);
            cp_wait();
            __syncthreads();
        }

        v8f accR, accI;
#pragma unroll
        for (int j = 0; j < 8; ++j) { accR[j] = bvr; accI[j] = bvi; }

#pragma unroll
        for (int kt = 0; kt < KT1; ++kt) {
            const int bofs = ((kt * 32 + ntg) * 32 + lane) * 2;
            Tile tbr, tbi;
            tbr.q[0] = Br[bofs]; tbr.q[1] = Br[bofs + 1];
            tbi.q[0] = Bi[bofs]; tbi.q[1] = Bi[bofs + 1];
            if (kt < 16) {   // reservoir K-section: complex A from LDS
                Tile ar, ai, tbn;
                const unsigned short* pR = shR + mrow * UU + kt * 32 + hi8;
                const unsigned short* pI = shI + mrow * UU + kt * 32 + hi8;
                ar.q[0] = *(const U4*)pR; ar.q[1] = *(const U4*)(pR + 16);
                ai.q[0] = *(const U4*)pI; ai.q[1] = *(const U4*)(pI + 16);
                tbn.q[0] = Bni[bofs]; tbn.q[1] = Bni[bofs + 1];
                accR = WMMA_BF16(ar.v, tbr.v, accR);
                accR = WMMA_BF16(ai.v, tbn.v, accR);
                accI = WMMA_BF16(ar.v, tbi.v, accI);
                accI = WMMA_BF16(ai.v, tbr.v, accI);
            } else {         // input K-section: real A (imag(x) == 0)
                Tile ax;
                const unsigned short* pX = shX + mrow * DD + (kt - 16) * 32 + hi8;
                ax.q[0] = *(const U4*)pX; ax.q[1] = *(const U4*)(pX + 16);
                accR = WMMA_BF16(ax.v, tbr.v, accR);
                accI = WMMA_BF16(ax.v, tbi.v, accI);
            }
        }

        // epilogue: complex tanh + leaky update, write bf16 state for step t+1
        unsigned short* oR = resR + (size_t)(t + 1) * (BB * UU);
        unsigned short* oI = resI + (size_t)(t + 1) * (BB * UU);
#pragma unroll
        for (int j = 0; j < 8; ++j) {
            int off = (mout + j) * UU + n;
            float a2 = fminf(fmaxf(2.f * accR[j], -30.f), 30.f);
            float b2 = 2.f * accI[j];
            float ex  = __expf(a2);
            float exi = 1.f / ex;
            float sh = 0.5f * (ex - exi), ch = 0.5f * (ex + exi);
            float sb = __sinf(b2), cb = __cosf(b2);
            float inv = 1.f / (ch + cb);
            float tr = sh * inv, ti = sb * inv;
            float hpR = bf2f(shR[off]), hpI = bf2f(shI[off]);   // h_{t-1} from LDS
            oR[off] = f2bf(0.2f * hpR + 0.8f * tr);
            oI[off] = f2bf(0.2f * hpI + 0.8f * ti);
        }

        // ---- grid barrier (cumulative counter) ----
        __threadfence();
        __syncthreads();
        if (tid == 0) {
            atomicAdd(ctr, 1u);
            const unsigned target = (unsigned)(t + 1) * NWG;
            while (__hip_atomic_load(ctr, __ATOMIC_RELAXED, __HIP_MEMORY_SCOPE_AGENT) < target)
                __builtin_amdgcn_s_sleep(1);
        }
        __syncthreads();
        __builtin_amdgcn_fence(__ATOMIC_ACQUIRE, "agent");
    }
}

// ---------------------------------------------------------------------------
// Readout: feats[b,t,:] @ W_ro, complex.  feats = [h_t (complex) | x_t | x_{t-1..t-3}] (real).
// 256 blocks x 256 threads; block -> (b, 64-t chunk); wave -> (m-tile, n-tile).
// Streaming GEMM with no cross-wave reuse -> no LDS staging needed.
// ---------------------------------------------------------------------------
__global__ void __launch_bounds__(256) wesn_readout(
    const unsigned* __restrict__ Bro,
    const unsigned short* __restrict__ resR, const unsigned short* __restrict__ resI,
    const unsigned short* __restrict__ xbf, float* __restrict__ out) {
    const int lane = threadIdx.x & 31;
    const int wave = threadIdx.x >> 5;
    const int mtl  = wave >> 1;            // 0..3
    const int ntg  = wave & 1;             // 0..1  (R = 32 -> 2 n-tiles)
    const int b    = blockIdx.x >> 3;
    const int t0   = ((blockIdx.x & 7) * 4 + mtl) * 16;
    const int hi8  = (lane & 16) >> 1;
    const int mrow = lane & 15;

    const U4* Ror  = (const U4*)Bro;
    const U4* Roi  = Ror + KTRO * 2 * 32 * 2;
    const U4* RoiN = Roi + KTRO * 2 * 32 * 2;

    v8f accR, accI;
#pragma unroll
    for (int j = 0; j < 8; ++j) { accR[j] = 0.f; accI[j] = 0.f; }

    const unsigned short* aR = resR + (size_t)(t0 + 1) * (BB * UU) + b * UU;
    const unsigned short* aI = resI + (size_t)(t0 + 1) * (BB * UU) + b * UU;

#pragma unroll
    for (int kt = 0; kt < 16; ++kt) {      // reservoir features (complex)
        Tile ar, ai, b0, b1, b2;
        const unsigned short* pR = aR + mrow * (BB * UU) + kt * 32 + hi8;
        const unsigned short* pI = aI + mrow * (BB * UU) + kt * 32 + hi8;
        ar.q[0] = *(const U4*)pR; ar.q[1] = *(const U4*)(pR + 16);
        ai.q[0] = *(const U4*)pI; ai.q[1] = *(const U4*)(pI + 16);
        const int bofs = ((kt * 2 + ntg) * 32 + lane) * 2;
        b0.q[0] = Ror[bofs];  b0.q[1] = Ror[bofs + 1];
        b1.q[0] = Roi[bofs];  b1.q[1] = Roi[bofs + 1];
        b2.q[0] = RoiN[bofs]; b2.q[1] = RoiN[bofs + 1];
        accR = WMMA_BF16(ar.v, b0.v, accR);
        accR = WMMA_BF16(ai.v, b2.v, accR);
        accI = WMMA_BF16(ar.v, b1.v, accI);
        accI = WMMA_BF16(ai.v, b0.v, accI);
    }
#pragma unroll
    for (int kt = 16; kt < KTRO; ++kt) {   // x + lag features (real)
        const int lag = (kt - 16) >> 1;
        const int dc  = ((kt - 16) & 1) * 32 + hi8;
        const unsigned short* pX =
            xbf + (size_t)(t0 + 3 - lag) * (BB * DD) + b * DD + mrow * (BB * DD) + dc;
        Tile ax, b0, b1;
        ax.q[0] = *(const U4*)pX; ax.q[1] = *(const U4*)(pX + 16);
        const int bofs = ((kt * 2 + ntg) * 32 + lane) * 2;
        b0.q[0] = Ror[bofs]; b0.q[1] = Ror[bofs + 1];
        b1.q[0] = Roi[bofs]; b1.q[1] = Roi[bofs + 1];
        accR = WMMA_BF16(ax.v, b0.v, accR);
        accI = WMMA_BF16(ax.v, b1.v, accI);
    }

    float2* o2 = (float2*)out;             // complex64 interleaved
#pragma unroll
    for (int j = 0; j < 8; ++j) {
        int t = t0 + j + ((lane & 16) ? 8 : 0);
        int r = ntg * 16 + (lane & 15);
        o2[((size_t)b * TT + t) * RR + r] = make_float2(accR[j], accI[j]);
    }
}

// ---------------------------------------------------------------------------
extern "C" void kernel_launch(void* const* d_in, const int* in_sizes, int n_in,
                              void* d_out, int out_size, void* d_ws, size_t ws_size,
                              hipStream_t stream) {
    const float* x   = (const float*)d_in[0];
    const float* kr  = (const float*)d_in[1];
    const float* ki  = (const float*)d_in[2];
    const float* rr  = (const float*)d_in[3];
    const float* ri  = (const float*)d_in[4];
    const float* br  = (const float*)d_in[5];
    const float* bi  = (const float*)d_in[6];
    const float* ror = (const float*)d_in[7];
    const float* roi = (const float*)d_in[8];

    char* w = (char*)d_ws;
    size_t off = 0;
    auto alloc = [&](size_t bytes) {
        void* p = w + off;
        off += (bytes + 255) & ~(size_t)255;
        return p;
    };
    unsigned*       ctr   = (unsigned*)alloc(256);
    unsigned*       Bscan = (unsigned*)alloc((size_t)NB_SCAN_DW * 4);
    unsigned*       BroS  = (unsigned*)alloc((size_t)NB_RO_DW * 4);
    unsigned short* xbf   = (unsigned short*)alloc((size_t)NX_ELEMS * 2);
    unsigned short* resR  = (unsigned short*)alloc((size_t)RES_ELEMS * 2);
    unsigned short* resI  = (unsigned short*)alloc((size_t)RES_ELEMS * 2);

    (void)hipMemsetAsync(ctr, 0, 256, stream);
    wesn_prep_scanB<<<(NB_SCAN_DW + 255) / 256, 256, 0, stream>>>(rr, ri, kr, ki, Bscan);
    wesn_prep_roB<<<(NB_RO_DW + 255) / 256, 256, 0, stream>>>(ror, roi, BroS);
    const int totp = NX_ELEMS + 2 * BB * UU;
    wesn_prep_x<<<(totp + 255) / 256, 256, 0, stream>>>(x, xbf, resR, resI);
    wesn_scan<<<NWG, 256, SMEM_BYTES, stream>>>(Bscan, xbf, resR, resI, br, bi, ctr);
    wesn_readout<<<256, 256, 0, stream>>>(BroS, resR, resI, xbf, (float*)d_out);
}
